// RoPEAttention_wMask_24172075942468
// MI455X (gfx1250) — matrix operand order repaired and checked by reading, not compiled, not run on hardware
//
#include <hip/hip_runtime.h>

// ---------------------------------------------------------------------------
// Types / WMMA helpers (CDNA5 gfx1250, wave32)
// ---------------------------------------------------------------------------
typedef __bf16 bf16_t;
typedef __bf16 v16bf __attribute__((ext_vector_type(16)));
typedef __bf16 v8bf  __attribute__((ext_vector_type(8)));
typedef float  v8f   __attribute__((ext_vector_type(8)));

static __device__ __forceinline__ v16bf cat8(v8bf lo, v8bf hi) {
  return __builtin_shufflevector(lo, hi, 0,1,2,3,4,5,6,7,8,9,10,11,12,13,14,15);
}

#define WMMA_BF16(A, B, C) \
  __builtin_amdgcn_wmma_f32_16x16x32_bf16(false, (A), false, (B), (short)0, (C), false, false)

// Async global->LDS 16B copy (GLOBAL_LOAD_ASYNC_TO_LDS_B128, GV mode).
// LDS address operand = low 32 bits of the generic LDS pointer.
static __device__ __forceinline__ void async_ld16(const bf16_t* g, bf16_t* l) {
  unsigned int loff = (unsigned int)(unsigned long long)l;
  asm volatile("global_load_async_to_lds_b128 %0, %1, off"
               :: "v"(loff), "v"(g) : "memory");
}
#define ASYNC_WAIT0() asm volatile("s_wait_asynccnt 0x0" ::: "memory")
#define ASYNC_WAIT4() asm volatile("s_wait_asynccnt 0x4" ::: "memory")

// Problem constants (from setup_inputs)
#define BATCH 8
#define MSEQ  2048
#define DIM   512
#define NHEAD 8
#define HDIM  64

// ---------------------------------------------------------------------------
// fp32 -> bf16 elementwise conversion
// ---------------------------------------------------------------------------
__global__ void cvt_f32_bf16(const float* __restrict__ in, bf16_t* __restrict__ out, int n) {
  int i = blockIdx.x * 256 + threadIdx.x;
  if (i < n) out[i] = (bf16_t)in[i];
}

// ---------------------------------------------------------------------------
// RoPE: read fp32 [B*M, D], rotate pairs, write bf16.
// theta_j = 10000^(-2*(j-1)/D)  (faithful to reference's (i-1) quirk)
// ---------------------------------------------------------------------------
__global__ void rope_f32_to_bf16(const float* __restrict__ t, bf16_t* __restrict__ out) {
  int i = blockIdx.x * 256 + threadIdx.x;          // over B*M*(D/2)
  const int dhalf = DIM / 2;
  int j   = i & (dhalf - 1);
  int row = i >> 8;                                 // dhalf == 256
  int pos = row & (MSEQ - 1);
  float theta = __expf((-2.0f * ((float)j - 1.0f) / (float)DIM) * 9.210340371976184f);
  float ang = (float)pos * theta;
  float c, s;
  __sincosf(ang, &s, &c);
  size_t base = (size_t)row * DIM + 2 * j;
  float e = t[base + 0];
  float o = t[base + 1];
  out[base + 0] = (bf16_t)( e * c + o * s);
  out[base + 1] = (bf16_t)(-e * s + o * c);
}

// ---------------------------------------------------------------------------
// GEMM: C[M,N] = A[M,K] * W[N,K]^T (+ bias[N])
// A, W bf16 row-major. Output fp32 (outF) or bf16 (outH).
// Block tile 128x128x32, 256 threads = 8 waves, wave tile 32x64 (2x4 frags).
// Double-buffered LDS; tiles staged by GLOBAL_LOAD_ASYNC_TO_LDS_B128 one tile
// ahead; ASYNCcnt in-order completion makes "wait <= 4" a previous-tile fence.
// LDS row stride padded to 48 bf16 (96B): 16B-aligned frag loads.
// ---------------------------------------------------------------------------
#define GBM 128
#define GBN 128
#define GBK 32
#define GLD 48

__global__ __launch_bounds__(256)
void gemm_bf16_wmma(const bf16_t* __restrict__ A,
                    const bf16_t* __restrict__ W,
                    const float*  __restrict__ bias,
                    float*        __restrict__ outF,
                    bf16_t*       __restrict__ outH,
                    int M, int N, int K) {
  __shared__ bf16_t sA[2][GBM * GLD];
  __shared__ bf16_t sW[2][GBN * GLD];

  const int tid  = threadIdx.x;
  const int wave = tid >> 5;
  const int lane = tid & 31;
  const int l16  = lane & 15;
  const int kh   = lane >> 4;

  const int bm = blockIdx.y * GBM;
  const int bn = blockIdx.x * GBN;

  const int wm = (wave >> 1) * 32;   // 0,32,64,96
  const int wn = (wave & 1) * 64;    // 0,64

  // per-thread tile-chunk mapping: 2 x (A chunk + W chunk) of 16B
  const int r0  = tid >> 2;          // rows 0..63   (pass 0)
  const int r1  = r0 + 64;           // rows 64..127 (pass 1)
  const int sg  = (tid & 3) * 8;     // 0,8,16,24 (bf16 elems)

  auto issueTile = [&](int k0, int buf) {
    async_ld16(A + (size_t)(bm + r0) * K + k0 + sg, &sA[buf][r0 * GLD + sg]);
    async_ld16(W + (size_t)(bn + r0) * K + k0 + sg, &sW[buf][r0 * GLD + sg]);
    async_ld16(A + (size_t)(bm + r1) * K + k0 + sg, &sA[buf][r1 * GLD + sg]);
    async_ld16(W + (size_t)(bn + r1) * K + k0 + sg, &sW[buf][r1 * GLD + sg]);
  };

  v8f acc[2][4];
#pragma unroll
  for (int i = 0; i < 2; ++i)
#pragma unroll
    for (int j = 0; j < 4; ++j) acc[i][j] = (v8f)0.0f;

  const int nk = K / GBK;
  issueTile(0, 0);

  for (int t = 0; t < nk; ++t) {
    const int buf = t & 1;
    if (t + 1 < nk) { issueTile((t + 1) * GBK, buf ^ 1); ASYNC_WAIT4(); }
    else            { ASYNC_WAIT0(); }
    __syncthreads();   // all waves' tile-t async writes visible

    v16bf afr[2], bfr[4];
#pragma unroll
    for (int i = 0; i < 2; ++i) {
      const bf16_t* base = &sA[buf][(wm + i * 16 + l16) * GLD + kh * 8];
      afr[i] = cat8(*(const v8bf*)base, *(const v8bf*)(base + 16));
    }
#pragma unroll
    for (int j = 0; j < 4; ++j) {
      const bf16_t* base = &sW[buf][(wn + j * 16 + l16) * GLD + kh * 8];
      bfr[j] = cat8(*(const v8bf*)base, *(const v8bf*)(base + 16));
    }
#pragma unroll
    for (int i = 0; i < 2; ++i)
#pragma unroll
      for (int j = 0; j < 4; ++j)
        acc[i][j] = WMMA_BF16(afr[i], bfr[j], acc[i][j]);

    __syncthreads();   // done reading buf before it is refilled at t+2
  }

  const bool f32out = (outF != nullptr);
#pragma unroll
  for (int i = 0; i < 2; ++i) {
#pragma unroll
    for (int j = 0; j < 4; ++j) {
      int col = bn + wn + j * 16 + l16;
      float bv = bias ? bias[col] : 0.0f;
#pragma unroll
      for (int r = 0; r < 8; ++r) {
        int rowg = bm + wm + i * 16 + kh * 8 + r;
        float v = acc[i][j][r] + bv;
        if (f32out) outF[(size_t)rowg * N + col] = v;
        else        outH[(size_t)rowg * N + col] = (bf16_t)v;
      }
    }
  }
}

// ---------------------------------------------------------------------------
// Causal flash attention per (b,h): Q,K,V bf16 in [B*M, D] with head slice
// cols [h*64, h*64+64). Block = 128 q rows, 8 waves x 16 rows. Key tiles of 64.
// Q/K tiles staged with async-to-LDS; next K/V tile prefetched into L2.
// S = Q*K^T and O += P*V via bf16 WMMA; online softmax with shfl_xor row
// reductions; P bounced through per-wave LDS to convert C-layout -> A-layout.
// ---------------------------------------------------------------------------
#define AQT 128
#define AKT 64
#define ALD 80   // LDS row stride (bf16) for Q/K/Vt/P: 160B, 16B-aligned frags

__global__ __launch_bounds__(256)
void attn_flash_wmma(const bf16_t* __restrict__ q,
                     const bf16_t* __restrict__ k,
                     const bf16_t* __restrict__ v,
                     bf16_t* __restrict__ ctx) {
  __shared__ bf16_t sQ[AQT * ALD];
  __shared__ bf16_t sK[AKT * ALD];
  __shared__ bf16_t sV[HDIM * ALD];      // transposed: [n][key]
  __shared__ bf16_t sP[8 * 16 * ALD];    // per-wave 16 x ALD scratch

  const int tid  = threadIdx.x;
  const int wave = tid >> 5;
  const int lane = tid & 31;
  const int l16  = lane & 15;
  const int kh   = lane >> 4;

  const int qt = blockIdx.x;             // q tile index
  const int bh = blockIdx.y;             // b*NHEAD + h
  const int b  = bh >> 3;
  const int h  = bh & 7;
  const size_t baseRow = (size_t)b * MSEQ;
  const int colh = h * HDIM;

  // Load Q tile: 128 rows x 64 bf16 (8 x 16B per row) -> 4 async chunks/thread
#pragma unroll
  for (int p = 0; p < 4; ++p) {
    int c = p * 256 + tid;
    int r = c >> 3, sg = (c & 7) * 8;
    async_ld16(q + (baseRow + qt * AQT + r) * DIM + colh + sg, sQ + r * ALD + sg);
  }
  ASYNC_WAIT0();
  __syncthreads();

  // Preload this wave's Q A-frags (rows wave*16 .. +16, K = head dim 64)
  v16bf qf[2];
#pragma unroll
  for (int kk = 0; kk < 2; ++kk) {
    const bf16_t* base = sQ + (wave * 16 + l16) * ALD + kk * 32 + kh * 8;
    qf[kk] = cat8(*(const v8bf*)base, *(const v8bf*)(base + 16));
  }

  v8f o[4];
#pragma unroll
  for (int j = 0; j < 4; ++j) o[j] = (v8f)0.0f;
  float mrow[8], lrow[8];
#pragma unroll
  for (int r = 0; r < 8; ++r) { mrow[r] = -3.0e38f; lrow[r] = 0.0f; }

  const float scale = 0.125f;            // 1/sqrt(64)
  const int nkt = 2 * (qt + 1);          // causal: keys up to (qt+1)*128
  const int qlo = qt * AQT + wave * 16;

  for (int kt = 0; kt < nkt; ++kt) {
    __syncthreads();
    // Stage K tile async; V tile through VGPRs with transpose scatter.
#pragma unroll
    for (int p = 0; p < 2; ++p) {
      int c = p * 256 + tid;
      int r = c >> 3, sg = (c & 7) * 8;
      const size_t grow = (baseRow + kt * AKT + r) * DIM + colh + sg;
      async_ld16(k + grow, sK + r * ALD + sg);
      v8bf vv = *(const v8bf*)(v + grow);
#pragma unroll
      for (int e = 0; e < 8; ++e) sV[(sg + e) * ALD + r] = vv[e];
      if (kt + 1 < nkt) {                // warm L2 for the next key tile
        __builtin_prefetch(k + grow + (size_t)AKT * DIM, 0, 1);
        __builtin_prefetch(v + grow + (size_t)AKT * DIM, 0, 1);
      }
    }
    ASYNC_WAIT0();
    __syncthreads();

    const bool need_mask = (kt * AKT + AKT - 1) > qlo;   // wave-uniform

    // ---- S = scale * Q*K^T, masked; track running row max ----
    float mloc[8];
#pragma unroll
    for (int r = 0; r < 8; ++r) mloc[r] = mrow[r];
    v8f s[4];
#pragma unroll
    for (int nf = 0; nf < 4; ++nf) {
      v8f a = (v8f)0.0f;
#pragma unroll
      for (int kk = 0; kk < 2; ++kk) {
        const bf16_t* base = sK + (nf * 16 + l16) * ALD + kk * 32 + kh * 8;
        v16bf bf = cat8(*(const v8bf*)base, *(const v8bf*)(base + 16));
        a = WMMA_BF16(qf[kk], bf, a);
      }
#pragma unroll
      for (int r = 0; r < 8; ++r) {
        float sv = a[r] * scale;
        if (need_mask) {
          int rowg = qlo + kh * 8 + r;
          int keyg = kt * AKT + nf * 16 + l16;
          if (keyg > rowg) sv = -1.0e30f;
        }
        a[r] = sv;
        mloc[r] = fmaxf(mloc[r], sv);
      }
      s[nf] = a;
    }
    // cross-lane row max (16 lanes share a row; xor 1,2,4,8 stays in half)
#pragma unroll
    for (int off = 1; off <= 8; off <<= 1)
#pragma unroll
      for (int r = 0; r < 8; ++r)
        mloc[r] = fmaxf(mloc[r], __shfl_xor(mloc[r], off, 32));

    float alpha[8], rsum[8];
#pragma unroll
    for (int r = 0; r < 8; ++r) { alpha[r] = __expf(mrow[r] - mloc[r]); rsum[r] = 0.0f; }

    // ---- P = exp(S - m_new); store to per-wave LDS in row-major ----
    bf16_t* pw = sP + wave * (16 * ALD);
#pragma unroll
    for (int nf = 0; nf < 4; ++nf) {
#pragma unroll
      for (int r = 0; r < 8; ++r) {
        float p = __expf(s[nf][r] - mloc[r]);
        rsum[r] += p;
        pw[(kh * 8 + r) * ALD + nf * 16 + l16] = (bf16_t)p;
      }
    }
#pragma unroll
    for (int off = 1; off <= 8; off <<= 1)
#pragma unroll
      for (int r = 0; r < 8; ++r)
        rsum[r] += __shfl_xor(rsum[r], off, 32);
#pragma unroll
    for (int r = 0; r < 8; ++r) {
      lrow[r] = lrow[r] * alpha[r] + rsum[r];
      mrow[r] = mloc[r];
    }
#pragma unroll
    for (int j = 0; j < 4; ++j)
#pragma unroll
      for (int r = 0; r < 8; ++r) o[j][r] *= alpha[r];

    // wave-private LDS RAW: LDS is in-order per wave; keep compiler honest
    asm volatile("s_wait_dscnt 0x0" ::: "memory");

    // ---- O += P * V ----
#pragma unroll
    for (int j = 0; j < 4; ++j) {
#pragma unroll
      for (int kk = 0; kk < 2; ++kk) {
        const bf16_t* pb = pw + l16 * ALD + kk * 32 + kh * 8;
        v16bf pf = cat8(*(const v8bf*)pb, *(const v8bf*)(pb + 16));
        const bf16_t* vb = sV + (j * 16 + l16) * ALD + kk * 32 + kh * 8;
        v16bf vf = cat8(*(const v8bf*)vb, *(const v8bf*)(vb + 16));
        o[j] = WMMA_BF16(pf, vf, o[j]);
      }
    }
  }

  // normalize + write ctx (bf16) back into [B*M, D] head slice
#pragma unroll
  for (int j = 0; j < 4; ++j) {
#pragma unroll
    for (int r = 0; r < 8; ++r) {
      int rowg = qt * AQT + wave * 16 + kh * 8 + r;
      float val = o[j][r] / lrow[r];
      ctx[(baseRow + rowg) * DIM + colh + j * 16 + l16] = (bf16_t)val;
    }
  }
}

// ---------------------------------------------------------------------------
// Host orchestration
// ---------------------------------------------------------------------------
extern "C" void kernel_launch(void* const* d_in, const int* in_sizes, int n_in,
                              void* d_out, int out_size, void* d_ws, size_t ws_size,
                              hipStream_t stream) {
  const float* x   = (const float*)d_in[0];
  const float* w_q = (const float*)d_in[1];
  const float* w_k = (const float*)d_in[2];
  const float* w_v = (const float*)d_in[3];
  const float* ipw = (const float*)d_in[4];
  const float* ipb = (const float*)d_in[5];
  const float* ow  = (const float*)d_in[6];
  const float* ob  = (const float*)d_in[7];

  const int M  = BATCH * MSEQ;           // 16384
  const int N  = DIM;                    // 512
  const int K  = DIM;                    // 512
  const size_t ND = (size_t)M * DIM;     // 8388608

  char* ws = (char*)d_ws;
  size_t off = 0;
  auto walloc = [&](size_t bytes) -> void* {
    void* p = ws + off;
    off = (off + bytes + 255) & ~(size_t)255;
    return p;
  };
  bf16_t* xh   = (bf16_t*)walloc(ND * 2);
  bf16_t* wqh  = (bf16_t*)walloc((size_t)DIM * DIM * 2);
  bf16_t* wkh  = (bf16_t*)walloc((size_t)DIM * DIM * 2);
  bf16_t* wvh  = (bf16_t*)walloc((size_t)DIM * DIM * 2);
  bf16_t* ipwh = (bf16_t*)walloc((size_t)3 * DIM * DIM * 2);
  bf16_t* owh  = (bf16_t*)walloc((size_t)DIM * DIM * 2);
  float*  tmpF = (float*)walloc(ND * 4);
  bf16_t* rq   = (bf16_t*)walloc(ND * 2);
  bf16_t* rk   = (bf16_t*)walloc(ND * 2);
  bf16_t* rv   = (bf16_t*)walloc(ND * 2);
  bf16_t* q2   = (bf16_t*)walloc(ND * 2);
  bf16_t* k2   = (bf16_t*)walloc(ND * 2);
  bf16_t* v2   = (bf16_t*)walloc(ND * 2);
  bf16_t* ctx  = (bf16_t*)walloc(ND * 2);

  dim3 blk(256);
  // convert activations + weights to bf16
  cvt_f32_bf16<<<(int)(ND / 256), blk, 0, stream>>>(x, xh, (int)ND);
  cvt_f32_bf16<<<DIM * DIM / 256, blk, 0, stream>>>(w_q, wqh, DIM * DIM);
  cvt_f32_bf16<<<DIM * DIM / 256, blk, 0, stream>>>(w_k, wkh, DIM * DIM);
  cvt_f32_bf16<<<DIM * DIM / 256, blk, 0, stream>>>(w_v, wvh, DIM * DIM);
  cvt_f32_bf16<<<3 * DIM * DIM / 256, blk, 0, stream>>>(ipw, ipwh, 3 * DIM * DIM);
  cvt_f32_bf16<<<DIM * DIM / 256, blk, 0, stream>>>(ow, owh, DIM * DIM);

  dim3 ggrid(N / GBN, M / GBM);          // 4 x 128
  const int ropeBlocks = (int)(ND / 2 / 256);

  // q = rope(x @ w_q^T) ... as bf16
  gemm_bf16_wmma<<<ggrid, blk, 0, stream>>>(xh, wqh, nullptr, tmpF, nullptr, M, N, K);
  rope_f32_to_bf16<<<ropeBlocks, blk, 0, stream>>>(tmpF, rq);
  gemm_bf16_wmma<<<ggrid, blk, 0, stream>>>(xh, wkh, nullptr, tmpF, nullptr, M, N, K);
  rope_f32_to_bf16<<<ropeBlocks, blk, 0, stream>>>(tmpF, rk);
  gemm_bf16_wmma<<<ggrid, blk, 0, stream>>>(xh, wvh, nullptr, tmpF, nullptr, M, N, K);
  rope_f32_to_bf16<<<ropeBlocks, blk, 0, stream>>>(tmpF, rv);

  // in_proj: q/k/v = rot @ w{q,k,v}i^T + b  (bf16 out)
  gemm_bf16_wmma<<<ggrid, blk, 0, stream>>>(rq, ipwh + 0 * DIM * DIM, ipb + 0 * DIM,
                                            nullptr, q2, M, N, K);
  gemm_bf16_wmma<<<ggrid, blk, 0, stream>>>(rk, ipwh + 1 * DIM * DIM, ipb + 1 * DIM,
                                            nullptr, k2, M, N, K);
  gemm_bf16_wmma<<<ggrid, blk, 0, stream>>>(rv, ipwh + 2 * DIM * DIM, ipb + 2 * DIM,
                                            nullptr, v2, M, N, K);

  // causal flash attention
  dim3 agrid(MSEQ / AQT, BATCH * NHEAD); // 16 x 64
  attn_flash_wmma<<<agrid, blk, 0, stream>>>(q2, k2, v2, ctx);

  // output projection (fp32 out)
  gemm_bf16_wmma<<<ggrid, blk, 0, stream>>>(ctx, owh, ob, (float*)d_out, nullptr, M, N, K);

  (void)in_sizes; (void)n_in; (void)out_size; (void)ws_size;
}
